// LIAFResBlock_4784593568444
// MI455X (gfx1250) — matrix operand order repaired and checked by reading, not compiled, not run on hardware
//
#include <hip/hip_runtime.h>
#include <hip/hip_bf16.h>

typedef __attribute__((ext_vector_type(16))) _Float16 v16h;
typedef __attribute__((ext_vector_type(8)))  _Float16 v8h;
typedef __attribute__((ext_vector_type(4)))  _Float16 v4h;
typedef __attribute__((ext_vector_type(8)))  float    v8f;

// Problem constants
#define BB    16
#define COUT  128
#define TT    8
#define HOUT  28
#define WOUT  28
#define NPIX  784            // HOUT*WOUT
#define CHW   6272           // TT*NPIX (per (b,c) plane of conv outputs)
#define NTOT  100352         // BB*TT*HOUT*WOUT
#define NELEM 12845056       // BB*COUT*TT*HOUT*WOUT

// ---------------------------------------------------------------- utilities
__global__ void zero_f(float* __restrict__ p, int n) {
    int i = blockIdx.x * 256 + threadIdx.x;
    if (i < n) p[i] = 0.f;
}

__global__ void to_half4(const float* __restrict__ in, _Float16* __restrict__ out, int n4) {
    int i = blockIdx.x * 256 + threadIdx.x;
    if (i < n4) {
        const float4 v = ((const float4*)in)[i];
        v4h h = {(_Float16)v.x, (_Float16)v.y, (_Float16)v.z, (_Float16)v.w};
        ((v4h*)out)[i] = h;
    }
}

// Pre-pack conv weights (O, Cin, KH, KW) fp32 -> f16 WMMA A-fragment order.
// GEMM K ordering: k = (kh*KW + kx)*CIN + ci   (spatial-major, channel-minor)
// Fragment layout: off = ((m_tile*KT + k_tile)*32 + lane)*16 + e
//   m = m_tile*16 + lane%16 ;  k = k_tile*32 + (e + 16*(lane/16))
__global__ void prepack_w(const float* __restrict__ w, _Float16* __restrict__ out,
                          int cin, int khw, int kwd, int total) {
    int idx = blockIdx.x * 256 + threadIdx.x;
    if (idx >= total) return;
    int e    = idx & 15;
    int lane = (idx >> 4) & 31;
    int kt_mt = idx >> 9;
    int KT = (cin * khw) / 32;
    int k_tile = kt_mt % KT;
    int m_tile = kt_mt / KT;
    int m = m_tile * 16 + (lane & 15);
    int k = k_tile * 32 + e + 16 * (lane >> 4);
    int ci = k % cin;
    int hw = k / cin;
    int kh = hw / kwd;
    int kx = hw % kwd;
    out[idx] = (_Float16)w[(m * cin + ci) * khw + kh * kwd + kx];
}

// -------------------------------------------------- implicit-GEMM conv (WMMA)
// D = W(128 x K) * im2col(src)(K x 100352). Block: 8 waves, tile 128M x 64N.
// K ordering (kh,kx)-major / ci-minor: each 32-wide K-tile has ONE (kh,kx),
// so the gather is one bounds test + one address + one 8-load clause per
// thread, and the 8 elements are e-contiguous -> ONE ds_store_b128.
// Software pipelined with double-buffered LDS: one barrier per K-step.
template <int CIN_, int KH, int KW, int STRIDE, int PAD, int HIN_, int WIN_>
__global__ __launch_bounds__(256) void conv_gemm(const _Float16* __restrict__ src,
                                                 const _Float16* __restrict__ wfrag,
                                                 float* __restrict__ y) {
    constexpr int K   = CIN_ * KH * KW;
    constexpr int KT  = K / 32;
    constexpr int HW  = HIN_ * WIN_;

    __shared__ _Float16 lds_b[2][2048];  // double buffer: 4 n-subtiles x 512

    const int tid    = threadIdx.x;
    const int lane   = tid & 31;
    const int wave   = tid >> 5;
    const int n_base = blockIdx.x * 64;

    // gather-thread coordinates: 64 pixels x 4 k-groups of 8 contiguous k
    const int n_local = tid & 63;
    const int kg      = tid >> 6;                    // k-range [kg*8, kg*8+8)
    const int n_g     = n_base + n_local;
    const int ow = n_g % WOUT;
    int tq = n_g / WOUT;
    const int oh = tq % HOUT; tq /= HOUT;
    const int t  = tq % TT;
    const int b  = tq / TT;
    const int ih0 = oh * STRIDE - PAD;
    const int iw0 = ow * STRIDE - PAD;
    const int bt  = (b * CIN_) * TT + t;            // + ci*TT -> plane index

    // one LDS slot per thread: frag row = (k/16)*16 | n%16, e0 = k0%16
    const int lds_off = (n_local >> 4) * 512 +
                        ((((kg >> 1) << 4) | (n_local & 15)) << 4) + ((kg & 1) << 3);

    // uniform (scalar) K-tile decomposition state for the NEXT gather
    int kh = 0, kx = 0, cb = 0;
    v8h regs;
    auto gather = [&]() {
        const int ih = ih0 + kh;
        const int iw = iw0 + kx;
        const bool valid = ((unsigned)ih < (unsigned)HIN_) &&
                           ((unsigned)iw < (unsigned)WIN_);
        const _Float16* p = src + (bt + (cb + kg * 8) * TT) * HW + ih * WIN_ + iw;
        v8h r = {};                                  // zeros for padded lanes
        if (valid) {                                 // ONE exec region, 8-load clause
#pragma unroll
            for (int j = 0; j < 8; ++j)             // ci stride 1 -> imm offsets
                r[j] = p[j * TT * HW];
        }
        regs = r;
        cb += 32;                                    // advance (uniform)
        if (cb == CIN_) { cb = 0; kx++; if (KW > 1 && kx == KW) { kx = 0; kh++; } }
    };
    auto store_tile = [&](int buf) {
        *(v8h*)&lds_b[buf][lds_off] = regs;          // one ds_store_b128
    };

    v8f acc[4] = {};

    gather();                                        // values for kt = 0
    store_tile(0);
    __syncthreads();
    for (int kt = 0; kt < KT; ++kt) {
        if (kt + 1 < KT) gather();                  // kt+1 loads in flight

        const v16h a = *(const v16h*)(wfrag + (((wave * KT + kt) * 32 + lane) << 4));
        if (kt + 1 < KT)
            __builtin_prefetch(wfrag + (((wave * KT + kt + 1) * 32 + lane) << 4), 0, 1);

        const _Float16* bbuf = lds_b[kt & 1];
#pragma unroll
        for (int ns = 0; ns < 4; ++ns) {
            const v16h bm = *(const v16h*)(bbuf + ns * 512 + lane * 16);
            acc[ns] = __builtin_amdgcn_wmma_f32_16x16x32_f16(
                false, a, false, bm, (short)0, acc[ns], false, false);
        }
        if (kt + 1 < KT) store_tile((kt + 1) & 1);  // other buffer: no conflict
        __syncthreads();
    }

    // epilogue: D layout -> VGPR r holds M = m0 + r, N = lane%16
    const int m0 = wave * 16 + ((lane >> 4) << 3);
#pragma unroll
    for (int ns = 0; ns < 4; ++ns) {
        const int n = n_base + ns * 16 + (lane & 15);
        const int ow2 = n % WOUT;
        int q = n / WOUT;
        const int oh2 = q % HOUT; q /= HOUT;
        const int t2 = q % TT;
        const int b2 = q / TT;
        const int base = (b2 * COUT + m0) * CHW + t2 * NPIX + oh2 * WOUT + ow2;
#pragma unroll
        for (int r = 0; r < 8; ++r)
            y[base + r * CHW] = acc[ns][r];
    }
}

// ---------------------------------------------------------- BN statistics
__global__ __launch_bounds__(256) void channel_stats(const float* __restrict__ y,
                                                     float* __restrict__ sums) {
    __shared__ float r1[256], r2[256];
    const int c = blockIdx.x >> 3, seg = blockIdx.x & 7;
    const int o0 = seg * NPIX;                       // slice one t-plane per seg
    float s = 0.f, s2 = 0.f;
    for (int b = 0; b < BB; ++b) {
        const float* p = y + (b * COUT + c) * CHW + o0;
        for (int i = (int)threadIdx.x; i < NPIX; i += 256) {
            const float v = p[i];
            s += v; s2 += v * v;
        }
    }
    r1[threadIdx.x] = s; r2[threadIdx.x] = s2;
    __syncthreads();
    for (int st = 128; st > 0; st >>= 1) {
        if ((int)threadIdx.x < st) {
            r1[threadIdx.x] += r1[threadIdx.x + st];
            r2[threadIdx.x] += r2[threadIdx.x + st];
        }
        __syncthreads();
    }
    if (threadIdx.x == 0) {
        atomicAdd(&sums[c], r1[0]);
        atomicAdd(&sums[COUT + c], r2[0]);
    }
}

__global__ void finalize_bn(const float* __restrict__ sums,
                            const float* __restrict__ g, const float* __restrict__ bta,
                            float* __restrict__ ss) {
    const int c = threadIdx.x;                 // 128 threads
    const float inv_n = 1.f / (float)NTOT;
    const float mean  = sums[c] * inv_n;
    const float var   = sums[COUT + c] * inv_n - mean * mean;
    const float scale = g[c] * rsqrtf(var + 1e-5f);
    ss[c]        = scale;
    ss[COUT + c] = bta[c] - mean * scale;
}

// -------------------------------------------- BN-apply + LIF scan -> spikes
__global__ __launch_bounds__(256) void bnlif(const float* __restrict__ y1,
                                             const float* __restrict__ ss,
                                             const float* __restrict__ decay,
                                             _Float16* __restrict__ s1) {
    const int i = blockIdx.x * 256 + threadIdx.x;       // over B*C*H*W
    if (i >= BB * COUT * NPIX) return;
    const int pix = i % NPIX;
    const int bc  = i / NPIX;
    const int c   = bc % COUT;
    const float scale = ss[c], shift = ss[COUT + c];
    const float d = 1.f / (1.f + __expf(-decay[0]));
    const int base = bc * CHW + pix;
    float mem = y1[base] * scale + shift;
    float s = mem > 0.5f ? 1.f : 0.f;
    s1[base] = (_Float16)s;
#pragma unroll
    for (int t = 1; t < TT; ++t) {
        const float v = y1[base + t * NPIX] * scale + shift;
        mem = mem * d * (1.f - s) + v;
        s = mem > 0.5f ? 1.f : 0.f;
        s1[base + t * NPIX] = (_Float16)s;
    }
}

// ------------- final: both BNs + residual + threshold + second LIF scan
__global__ __launch_bounds__(256) void final_lif(const float* __restrict__ y2,
                                                 const float* __restrict__ ysc,
                                                 const float* __restrict__ ss2,
                                                 const float* __restrict__ ss3,
                                                 const float* __restrict__ decay,
                                                 float* __restrict__ out) {
    const int i = blockIdx.x * 256 + threadIdx.x;
    if (i >= BB * COUT * NPIX) return;
    const int pix = i % NPIX;
    const int bc  = i / NPIX;
    const int c   = bc % COUT;
    const float a2 = ss2[c], b2 = ss2[COUT + c];
    const float a3 = ss3[c], b3 = ss3[COUT + c];
    const float d = 1.f / (1.f + __expf(-decay[0]));
    const int base = bc * CHW + pix;
    float u = (y2[base] * a2 + b2 + ysc[base] * a3 + b3) > 0.5f ? 1.f : 0.f;
    float mem = u;
    float s = mem > 0.5f ? 1.f : 0.f;
    out[base] = s;
#pragma unroll
    for (int t = 1; t < TT; ++t) {
        const float ut = (y2[base + t * NPIX] * a2 + b2 +
                          ysc[base + t * NPIX] * a3 + b3) > 0.5f ? 1.f : 0.f;
        mem = mem * d * (1.f - s) + ut;
        s = mem > 0.5f ? 1.f : 0.f;
        out[base + t * NPIX] = s;
    }
}

// -------------------------------------------------------------------- launch
extern "C" void kernel_launch(void* const* d_in, const int* in_sizes, int n_in,
                              void* d_out, int out_size, void* d_ws, size_t ws_size,
                              hipStream_t stream) {
    (void)in_sizes; (void)n_in; (void)out_size; (void)ws_size;
    const float* x     = (const float*)d_in[0];
    const float* cv1_w = (const float*)d_in[1];
    const float* cv2_w = (const float*)d_in[3];
    const float* sc_w  = (const float*)d_in[5];
    const float* bn1_g = (const float*)d_in[7];
    const float* bn1_b = (const float*)d_in[8];
    const float* bn2_g = (const float*)d_in[9];
    const float* bn2_b = (const float*)d_in[10];
    const float* scn_g = (const float*)d_in[11];
    const float* scn_b = (const float*)d_in[12];
    const float* decay = (const float*)d_in[13];
    float* out = (float*)d_out;

    // Workspace layout (256B aligned). Total ~180 MB — fits MI455X's 192MB L2.
    char* ws = (char*)d_ws;
    size_t off = 0;
    auto take = [&](size_t bytes) -> char* {
        char* p = ws + off;
        off = (off + bytes + 255) & ~(size_t)255;
        return p;
    };
    const size_t NX = 25690112;  // B*Cin*T*56*56
    _Float16* xh  = (_Float16*)take(NX * 2);
    _Float16* w1f = (_Float16*)take((size_t)73728 * 2);   // 128 x 576
    _Float16* w2f = (_Float16*)take((size_t)147456 * 2);  // 128 x 1152
    _Float16* scf = (_Float16*)take((size_t)8192 * 2);    // 128 x 64
    float*    y1  = (float*)take((size_t)NELEM * 4);      // cv1 pre-BN; reused as ysc
    float*    y2  = (float*)take((size_t)NELEM * 4);
    _Float16* s1  = (_Float16*)take((size_t)NELEM * 2);   // spikes (f16)
    float*    sums = (float*)take(768 * 4);               // 3 convs x {sum,sumsq}[128]
    float*    ss   = (float*)take(768 * 4);               // 3 convs x {scale,shift}[128]
    float*    ysc  = y1;                                  // alias: y1 dead after bnlif

    zero_f<<<3, 256, 0, stream>>>(sums, 768);
    to_half4<<<(int)(NX / 1024), 256, 0, stream>>>(x, xh, (int)(NX / 4));
    prepack_w<<<288, 256, 0, stream>>>(cv1_w, w1f, 64, 9, 3, 73728);
    prepack_w<<<576, 256, 0, stream>>>(cv2_w, w2f, 128, 9, 3, 147456);
    prepack_w<<<32, 256, 0, stream>>>(sc_w, scf, 64, 1, 1, 8192);

    // cv1: 3x3 stride2 pad1 on x(56x56)
    conv_gemm<64, 3, 3, 2, 1, 56, 56><<<1568, 256, 0, stream>>>(xh, w1f, y1);
    channel_stats<<<1024, 256, 0, stream>>>(y1, sums + 0);
    finalize_bn<<<1, 128, 0, stream>>>(sums + 0, bn1_g, bn1_b, ss + 0);
    bnlif<<<6272, 256, 0, stream>>>(y1, ss + 0, decay, s1);

    // cv2: 3x3 stride1 pad1 on spikes(28x28)
    conv_gemm<128, 3, 3, 1, 1, 28, 28><<<1568, 256, 0, stream>>>(s1, w2f, y2);
    // shortcut: 1x1 stride2 on x(56x56) -> ysc (aliases y1)
    conv_gemm<64, 1, 1, 2, 0, 56, 56><<<1568, 256, 0, stream>>>(xh, scf, ysc);

    channel_stats<<<1024, 256, 0, stream>>>(y2, sums + 256);
    channel_stats<<<1024, 256, 0, stream>>>(ysc, sums + 512);
    finalize_bn<<<1, 128, 0, stream>>>(sums + 256, bn2_g, bn2_b, ss + 256);
    finalize_bn<<<1, 128, 0, stream>>>(sums + 512, scn_g, scn_b, ss + 512);

    final_lif<<<6272, 256, 0, stream>>>(y2, ysc, ss + 256, ss + 512, decay, out);
}
